// EALSTM_13005160972908
// MI455X (gfx1250) — matrix operand order repaired and checked by reading, not compiled, not run on hardware
//
#include <hip/hip_runtime.h>

// EA-LSTM fused scan for MI455X (gfx1250, wave32).
// Grid: 8 workgroups (one per 16-row batch tile), 512 threads = 16 waves.
// Wave w owns H columns [16w, 16w+16). Per step: 28 v_wmma_f32_16x16x32_bf16.

#define Bv    128
#define Tv    2048
#define Hv    256
#define WPAD  264   // padded LDS row (elements) for weights  (528B rows, 16B aligned)
#define HPAD  264   // padded LDS row for h tile
#define XP    32    // x tile row = DYN+STAT = 32 features = one WMMA K slice

typedef __attribute__((ext_vector_type(16))) __bf16        v16bf;
typedef __attribute__((ext_vector_type(8)))  float          v8f;
typedef __attribute__((ext_vector_type(16))) unsigned short v16u16;
typedef __attribute__((ext_vector_type(8)))  unsigned short v8u16;

struct Frag2x8 { v8u16 lo, hi; };   // 32B, two 16B halves

__device__ __forceinline__ unsigned short f2bf(float f) {
  unsigned int u = __float_as_uint(f);
  u += 0x7FFFu + ((u >> 16) & 1u);          // round-to-nearest-even
  return (unsigned short)(u >> 16);
}
__device__ __forceinline__ float sigmoid_(float x) {
  return __fdividef(1.f, 1.f + __expf(-x));
}
__device__ __forceinline__ float tanh_(float x) {
  return 1.f - __fdividef(2.f, __expf(2.f * x) + 1.f);
}
// A fragment (16x32 bf16): lane m=lane&15; 16 contiguous K values starting at
// kstep*32 + (lane<16 ? 0 : 16)           -> two 16B LDS loads, gap = 8 elements.
// B fragment (32x16 bf16): lane n=lane&15; 16 contiguous K values starting at
// kstep*32 + (lane<16 ? 0 : 16)           -> two 16B LDS loads, gap = 8 elements.
__device__ __forceinline__ v16bf ldfrag(const unsigned short* p, int gap) {
  Frag2x8 f;
  f.lo = *(const v8u16*)(p);
  f.hi = *(const v8u16*)(p + gap);
  return __builtin_bit_cast(v16bf, f);
}
#define WMMA_BF16(a, b, c) \
  __builtin_amdgcn_wmma_f32_16x16x32_bf16(false, (a), false, (b), (short)0, (c), false, false)

__global__ __launch_bounds__(512)
void ealstm_scan(const float* __restrict__ x,    const float* __restrict__ c0,
                 const float* __restrict__ h0,   const float* __restrict__ Wi,
                 const float* __restrict__ bi,   const float* __restrict__ Wf_x,
                 const float* __restrict__ bf_x, const float* __restrict__ Wf_h,
                 const float* __restrict__ bf_h, const float* __restrict__ Wg_x,
                 const float* __restrict__ bg_x, const float* __restrict__ Wg_h,
                 const float* __restrict__ bg_h, const float* __restrict__ Wo_x,
                 const float* __restrict__ bo_x, const float* __restrict__ Wo_h,
                 const float* __restrict__ bo_h, const float* __restrict__ Wout,
                 const float* __restrict__ bout, float* __restrict__ out)
{
  // ~283 KB LDS per workgroup (one WG per WGP; gfx1250 allows 320 KB).
  __shared__ unsigned short wfL[256 * WPAD];      // Wf_h bf16, row-major, padded
  __shared__ unsigned short wgL[256 * WPAD];      // Wg_h bf16
  __shared__ unsigned short hb[2][16 * HPAD];     // h tile (16 x 256 bf16), ping-pong
  __shared__ unsigned short xb[2][16 * XP];       // x tile (16 x 32 bf16), ping-pong
  __shared__ float oacc[2][16];                   // out-projection accumulators

  const int tid  = threadIdx.x;
  const int lane = tid & 31;
  const int wid  = tid >> 5;          // wave id = H tile, 0..15
  const int nloc = lane & 15;
  const int half = lane >> 4;         // 0: lanes 0-15, 1: lanes 16-31
  const int n    = wid * 16 + nloc;   // global H index of this lane's column
  const int b0   = blockIdx.x * 16;   // batch tile base

  // ---- stage recurrent f/g weights into LDS as bf16 ----
  for (int idx = tid; idx < 256 * 256; idx += 512) {
    int r = idx >> 8, k = idx & 255;
    wfL[r * WPAD + k] = f2bf(Wf_h[idx]);
    wgL[r * WPAD + k] = f2bf(Wg_h[idx]);
  }
  // ---- h0 broadcast over the 16 batch rows ----
  for (int idx = tid; idx < 16 * 256; idx += 512) {
    int m = idx >> 8, k = idx & 255;
    hb[0][m * HPAD + k] = f2bf(h0[k]);
  }
  // ---- x tile for t = 0 ----
  {
    int m = tid >> 5, c = lane;
    xb[0][m * XP + c] = f2bf(x[((size_t)(b0 + m) * Tv + 0) * 32 + c]);
  }
  if (tid < 32) oacc[tid >> 4][tid & 15] = 0.f;

  // ---- per-wave VGPR-resident fragments ----
  // Wo_h B-fragments: 8 K-steps of 16 contiguous bf16 each (64 VGPRs/lane).
  v16bf wofr[8];
  #pragma unroll
  for (int ks = 0; ks < 8; ++ks) {
    v16u16 u;
    const float* src = Wo_h + (size_t)n * 256 + ks * 32 + half * 16;
    #pragma unroll
    for (int j = 0; j < 16; ++j) u[j] = f2bf(src[j]);
    wofr[ks] = __builtin_bit_cast(v16bf, u);
  }
  // x-path B-fragments (K=32 = [8 dyn | 24 stat]); dyn gates use K<8, input gate K>=8.
  v16bf wxf, wxg, wxo, wxi;
  {
    v16u16 uf, ug, uo, ui;
    #pragma unroll
    for (int j = 0; j < 16; ++j) {
      int k = half * 16 + j;
      uf[j] = (k < 8) ? f2bf(Wf_x[n * 8 + k]) : (unsigned short)0;
      ug[j] = (k < 8) ? f2bf(Wg_x[n * 8 + k]) : (unsigned short)0;
      uo[j] = (k < 8) ? f2bf(Wo_x[n * 8 + k]) : (unsigned short)0;
      ui[j] = (k >= 8) ? f2bf(Wi[n * 24 + (k - 8)]) : (unsigned short)0;
    }
    wxf = __builtin_bit_cast(v16bf, uf);
    wxg = __builtin_bit_cast(v16bf, ug);
    wxo = __builtin_bit_cast(v16bf, uo);
    wxi = __builtin_bit_cast(v16bf, ui);
  }
  const float biasf = bf_x[n] + bf_h[n];
  const float biasg = bg_x[n] + bg_h[n];
  const float biaso = bo_x[n] + bo_h[n];
  const float biasi = bi[n];
  const float woutn = Wout[n];
  const float boutv = bout[0];

  v8f cc;                              // persistent c state: (m = half*8+r, n)
  { float cv = c0[n];
    #pragma unroll
    for (int r = 0; r < 8; ++r) cc[r] = cv; }

  __syncthreads();

  for (int t = 0; t < Tv; ++t) {
    const int p = t & 1, np = p ^ 1;

    // prefetch / early-load next x tile (hides HBM->L2 latency under WMMAs)
    float xnext = 0.f;
    {
      int m = tid >> 5, c = lane;
      if (t + 1 < Tv) xnext = x[((size_t)(b0 + m) * Tv + (t + 1)) * 32 + c];
      if (t + 8 < Tv)
        __builtin_prefetch(&x[((size_t)(b0 + m) * Tv + (t + 8)) * 32 + c], 0, 1);
    }
    // drain previous step's output projection (double-buffered, race-free)
    if (t > 0 && wid == 0 && lane < 16) {
      float v = oacc[np][lane];
      out[(size_t)(b0 + lane) * Tv + (t - 1)] = v + boutv;
      oacc[np][lane] = 0.f;
    }

    // seed accumulators with fused biases
    v8f accf, accg, acco, acci;
    #pragma unroll
    for (int r = 0; r < 8; ++r) {
      accf[r] = biasf; accg[r] = biasg; acco[r] = biaso; acci[r] = biasi;
    }

    // x contribution: one WMMA per gate (K = 32 features)
    {
      v16bf ax = ldfrag(&xb[p][nloc * XP + half * 16], 8);
      accf = WMMA_BF16(ax, wxf, accf);
      accg = WMMA_BF16(ax, wxg, accg);
      acco = WMMA_BF16(ax, wxo, acco);
      acci = WMMA_BF16(ax, wxi, acci);
    }
    // recurrent contribution: K = 256 in 8 slices, A(h) shared across 3 gates
    #pragma unroll
    for (int ks = 0; ks < 8; ++ks) {
      v16bf ah = ldfrag(&hb[p][nloc * HPAD + ks * 32 + half * 16], 8);
      accf = WMMA_BF16(ah, ldfrag(&wfL[n * WPAD + ks * 32 + half * 16], 8), accf);
      accg = WMMA_BF16(ah, ldfrag(&wgL[n * WPAD + ks * 32 + half * 16], 8), accg);
      acco = WMMA_BF16(ah, wofr[ks], acco);
    }

    // gates, out projection, c/h update
    #pragma unroll
    for (int r = 0; r < 8; ++r) {
      float f = sigmoid_(accf[r]);
      float g = tanh_(accg[r]);
      float o = sigmoid_(acco[r]);
      float i = sigmoid_(acci[r]);
      // out = o @ Wout^T: reduce over this wave's 16 H-columns, then LDS atomic
      float s = o * woutn;
      s += __shfl_xor(s, 1, 16);
      s += __shfl_xor(s, 2, 16);
      s += __shfl_xor(s, 4, 16);
      s += __shfl_xor(s, 8, 16);
      if (nloc == 0) atomicAdd(&oacc[p][half * 8 + r], s);
      float c = f * cc[r] + i * g;
      cc[r] = c;
      float h = o * tanh_(c);
      hb[np][(half * 8 + r) * HPAD + n] = f2bf(h);   // publish h_{t} for step t+1
    }
    // stage next x tile into the ping-pong buffer
    {
      int m = tid >> 5, c = lane;
      xb[np][m * XP + c] = f2bf(xnext);
    }
    __syncthreads();   // one barrier per timestep
  }
  // flush final timestep's output
  if (wid == 0 && lane < 16)
    out[(size_t)(b0 + lane) * Tv + (Tv - 1)] = oacc[(Tv - 1) & 1][lane] + boutv;
}

extern "C" void kernel_launch(void* const* d_in, const int* in_sizes, int n_in,
                              void* d_out, int out_size, void* d_ws, size_t ws_size,
                              hipStream_t stream) {
  const float* x    = (const float*)d_in[0];
  const float* c0   = (const float*)d_in[1];
  const float* h0   = (const float*)d_in[2];
  const float* Wi   = (const float*)d_in[3];
  const float* bi   = (const float*)d_in[4];
  const float* Wf_x = (const float*)d_in[5];
  const float* bf_x = (const float*)d_in[6];
  const float* Wf_h = (const float*)d_in[7];
  const float* bf_h = (const float*)d_in[8];
  const float* Wg_x = (const float*)d_in[9];
  const float* bg_x = (const float*)d_in[10];
  const float* Wg_h = (const float*)d_in[11];
  const float* bg_h = (const float*)d_in[12];
  const float* Wo_x = (const float*)d_in[13];
  const float* bo_x = (const float*)d_in[14];
  const float* Wo_h = (const float*)d_in[15];
  const float* bo_h = (const float*)d_in[16];
  const float* Wout = (const float*)d_in[17];
  const float* bout = (const float*)d_in[18];
  float* outp = (float*)d_out;

  dim3 grid(Bv / 16), block(512);
  ealstm_scan<<<grid, block, 0, stream>>>(x, c0, h0, Wi, bi, Wf_x, bf_x, Wf_h, bf_h,
                                          Wg_x, bg_x, Wg_h, bg_h, Wo_x, bo_x, Wo_h,
                                          bo_h, Wout, bout, outp);
}
